// HGT_23055384445694
// MI455X (gfx1250) — compile-verified, matrix operand order checked by reading, not run on hardware
//
#include <hip/hip_runtime.h>
#include <hip/hip_bf16.h>

// ---------------- problem constants (match reference) ----------------
#define NA    30000
#define NPP   50000
#define CDIM  256
#define HH    8
#define DD    32
#define LL    2
#define TT    2
#define RR    3
#define OUTC  64
#define FA    128
#define FP    256
#define SCALE 0.17677669529663687f   // 1/sqrt(32)

#define AS1 __attribute__((address_space(1)))
#define AS3 __attribute__((address_space(3)))

#if defined(__has_builtin)
#if __has_builtin(__builtin_amdgcn_global_load_async_to_lds_b128)
#define HAVE_ASYNC_LDS 1
#endif
#endif
#ifndef HAVE_ASYNC_LDS
#define HAVE_ASYNC_LDS 0
#endif

// ---------------- WMMA types ----------------
typedef __attribute__((ext_vector_type(16))) __bf16 v16bf;
typedef __attribute__((ext_vector_type(8)))  float  v8f;
typedef __attribute__((ext_vector_type(4)))  int    i32x4;

union Frag {
    v16bf v;
    uint4 u2[2];
    unsigned short h[16];
};

__device__ __forceinline__ v8f wmma_bf16(v16bf a, v16bf b, v8f c) {
    return __builtin_amdgcn_wmma_f32_16x16x32_bf16(
        /*neg_a=*/false, a, /*neg_b=*/false, b,
        /*c_mod=*/(short)0, c, /*reuse_a=*/false, /*reuse_b=*/false);
}

__device__ __forceinline__ unsigned short f2bf(float x) {
    unsigned u = __float_as_uint(x);
    unsigned r = u + 0x7FFFu + ((u >> 16) & 1u);   // round to nearest even
    return (unsigned short)(r >> 16);
}

#if HAVE_ASYNC_LDS
// 16-byte async copy global -> LDS (tracked by ASYNCcnt)
__device__ __forceinline__ void async_b128(const void* gsrc, void* ldst) {
    i32x4* gg = (i32x4*)const_cast<void*>(gsrc);   // drop const / retype (generic AS)
    i32x4* ll = (i32x4*)ldst;
    __builtin_amdgcn_global_load_async_to_lds_b128(
        (AS1 i32x4*)gg,    // addrspacecast generic -> global
        (AS3 i32x4*)ll,    // addrspacecast generic -> local
        0, 0);
}
#endif

// ---------------- elementwise helpers ----------------
__global__ void cvt_bf16_kernel(const float* __restrict__ in,
                                unsigned short* __restrict__ out, long n) {
    long i = (long)blockIdx.x * blockDim.x + threadIdx.x;
    if (i < n) out[i] = f2bf(in[i]);
}

__global__ void gelu_to_bf16_kernel(const float* __restrict__ in,
                                    unsigned short* __restrict__ out, long n) {
    long i = (long)blockIdx.x * blockDim.x + threadIdx.x;
    if (i < n) {
        float x = in[i];
        float t = tanhf(0.7978845608f * (x + 0.044715f * x * x * x));
        out[i] = f2bf(0.5f * x * (1.0f + t));
    }
}

// pack weight f32 [K][N] -> bf16 [N][K]
__global__ void pack_wt_kernel(const float* __restrict__ in,
                               unsigned short* __restrict__ out, int K, int N) {
    long id = (long)blockIdx.x * blockDim.x + threadIdx.x;
    if (id >= (long)K * N) return;
    int n = (int)(id / K), k = (int)(id % K);
    out[id] = f2bf(in[(long)k * N + n]);
}

// pack head weights f32 [blocks][32(d)][32(e)] -> bf16 [blocks][32(e)][32(d)]
__global__ void pack_head_kernel(const float* __restrict__ in,
                                 unsigned short* __restrict__ out, int blocks) {
    long id = (long)blockIdx.x * blockDim.x + threadIdx.x;
    if (id >= (long)blocks * 1024) return;
    long blk = id >> 10;
    int rem = (int)(id & 1023), e = rem >> 5, d = rem & 31;
    out[id] = f2bf(in[blk * 1024 + d * 32 + e]);
}

// ---------------- WMMA GEMM: out = epi(A[M x K] * Bt^T + bias) ----------------
// A: bf16 row-major [M][K]; Bt: bf16 packed [N][K]; BK=32 per WMMA step.
#define BM  128
#define BN  128
#define BK  32
#define BKP 40   // padded LDS row (conflict-free ds_load_b128)

// Out-of-range rows are CLAMPED (not zero-filled): a garbage A-row r only
// feeds output row r (never stored) and a garbage B-row n only feeds output
// column n (never stored), so clamping is numerically safe for kept outputs.
#if HAVE_ASYNC_LDS
__device__ __forceinline__ void stage_issue_async(
    unsigned short (&As)[2][BM][BKP], unsigned short (&Bs)[2][BN][BKP],
    int s, const unsigned short* __restrict__ A, const unsigned short* __restrict__ Bt,
    long gra, long gbb, int K, int kt, int row, int c16) {
    const unsigned short* ap = A + gra * (long)K + (long)kt * BK + c16;
    const unsigned short* bp = Bt + gbb * (long)K + (long)kt * BK + c16;
    async_b128(ap,     &As[s][row][c16]);
    async_b128(ap + 8, &As[s][row][c16 + 8]);
    async_b128(bp,     &Bs[s][row][c16]);
    async_b128(bp + 8, &Bs[s][row][c16 + 8]);
}
#else
__device__ __forceinline__ void stage_load_sync(
    unsigned short (&As)[2][BM][BKP], unsigned short (&Bs)[2][BN][BKP],
    int s, const unsigned short* __restrict__ A, const unsigned short* __restrict__ Bt,
    long gra, long gbb, int K, int kt, int row, int c16) {
    const uint4* ap = (const uint4*)(A + gra * (long)K + (long)kt * BK + c16);
    uint4 a0 = ap[0], a1 = ap[1];
    *(uint4*)&As[s][row][c16]     = a0;
    *(uint4*)&As[s][row][c16 + 8] = a1;
    const uint4* bp = (const uint4*)(Bt + gbb * (long)K + (long)kt * BK + c16);
    uint4 b0 = bp[0], b1 = bp[1];
    *(uint4*)&Bs[s][row][c16]     = b0;
    *(uint4*)&Bs[s][row][c16 + 8] = b1;
}
#endif

// MODE: 1 = relu(acc+bias) -> f32 + bf16
//       2 = acc+bias       -> f32
//       3 = sigmoid-skip blend: v = b*(acc+bias) + (1-b)*xold -> f32 + bf16
template <int MODE>
__global__ __launch_bounds__(256) void wmma_gemm_kernel(
    const unsigned short* __restrict__ A, const unsigned short* __restrict__ Bt,
    const float* __restrict__ bias, int M, int N, int K,
    float* __restrict__ outF, unsigned short* __restrict__ outBF,
    const float* __restrict__ xold, const float* __restrict__ skipArr, int skipIdx) {
    __shared__ unsigned short As[2][BM][BKP];
    __shared__ unsigned short Bs[2][BN][BKP];

    const int tid  = threadIdx.x;
    const int lane = tid & 31;
    const int wave = tid >> 5;          // 8 waves
    const int wm   = wave >> 1;         // 0..3  (32-row strip)
    const int wn   = wave & 1;          // 0..1  (64-col strip)
    const int lrow = lane & 15;
    const int khalf = (lane >> 4) << 3; // 0 or 8

    const int row = tid >> 1;           // 0..127 tile row
    const int c16 = (tid & 1) << 4;     // 0 or 16
    const long gr = (long)blockIdx.y * BM + row;
    const long gb = (long)blockIdx.x * BN + row;
    const long gra = (gr < M) ? gr : (long)(M - 1);   // clamp (see note above)
    const long gbb = (gb < N) ? gb : (long)(N - 1);

    v8f zero = {};
    v8f acc[2][4];
#pragma unroll
    for (int i = 0; i < 2; ++i)
#pragma unroll
        for (int j = 0; j < 4; ++j) acc[i][j] = zero;

    const int KT = K / BK;

#if HAVE_ASYNC_LDS
    stage_issue_async(As, Bs, 0, A, Bt, gra, gbb, K, 0, row, c16);
    for (int kt = 0; kt < KT; ++kt) {
        const int cur = kt & 1;
        if (kt + 1 < KT) {
            stage_issue_async(As, Bs, cur ^ 1, A, Bt, gra, gbb, K, kt + 1, row, c16);
            asm volatile("s_wait_asynccnt 0x4" ::: "memory");  // cur's 4 loads done
        } else {
            asm volatile("s_wait_asynccnt 0x0" ::: "memory");
        }
        __syncthreads();   // all waves' cur-stage loads visible
#else
    stage_load_sync(As, Bs, 0, A, Bt, gra, gbb, K, 0, row, c16);
    __syncthreads();
    for (int kt = 0; kt < KT; ++kt) {
        const int cur = kt & 1;
        if (kt + 1 < KT) {
            stage_load_sync(As, Bs, cur ^ 1, A, Bt, gra, gbb, K, kt + 1, row, c16);
            if (kt + 2 < KT)
                __builtin_prefetch(A + gra * (long)K + (long)(kt + 2) * BK, 0, 1);
        }
#endif
        Frag af[2], bfr[4];
#pragma unroll
        for (int mi = 0; mi < 2; ++mi) {
            int r = wm * 32 + mi * 16 + lrow;
            af[mi].u2[0] = *(const uint4*)&As[cur][r][khalf];
            af[mi].u2[1] = *(const uint4*)&As[cur][r][khalf + 16];
        }
#pragma unroll
        for (int ni = 0; ni < 4; ++ni) {
            int n = wn * 64 + ni * 16 + lrow;
            bfr[ni].u2[0] = *(const uint4*)&Bs[cur][n][khalf];
            bfr[ni].u2[1] = *(const uint4*)&Bs[cur][n][khalf + 16];
        }
#pragma unroll
        for (int mi = 0; mi < 2; ++mi)
#pragma unroll
            for (int ni = 0; ni < 4; ++ni)
                acc[mi][ni] = wmma_bf16(af[mi].v, bfr[ni].v, acc[mi][ni]);
        __syncthreads();   // protect cur stage before it is refilled
    }

    float beta = 0.f, ombeta = 0.f;
    if (MODE == 3) {
        float sv = skipArr[skipIdx];
        beta = 1.0f / (1.0f + __expf(-sv));
        ombeta = 1.0f - beta;
    }
    const int rb = (lane >> 4) << 3;    // lanes 16-31 hold rows M+8..M+15
    const long rowBase = (long)blockIdx.y * BM + wm * 32 + rb;
    const int  colBase = blockIdx.x * BN + wn * 64 + lrow;
    const bool fullTile = ((long)blockIdx.y * BM + BM <= M) && (blockIdx.x * BN + BN <= N);
    if (fullTile) {
#pragma unroll
        for (int mi = 0; mi < 2; ++mi) {
#pragma unroll
            for (int ni = 0; ni < 4; ++ni) {
                int gcol = colBase + ni * 16;
                float bv = bias[gcol];
#pragma unroll
                for (int j = 0; j < 8; ++j) {
                    long grow = rowBase + mi * 16 + j;
                    long idx = grow * (long)N + gcol;
                    float v = acc[mi][ni][j] + bv;
                    if (MODE == 1) v = fmaxf(v, 0.0f);
                    if (MODE == 3) v = beta * v + ombeta * xold[idx];
                    outF[idx] = v;
                    if (MODE == 1 || MODE == 3) outBF[idx] = f2bf(v);
                }
            }
        }
    } else {
#pragma unroll
        for (int mi = 0; mi < 2; ++mi) {
#pragma unroll
            for (int ni = 0; ni < 4; ++ni) {
                int gcol = colBase + ni * 16;
                if (gcol >= N) continue;
                float bv = bias[gcol];
#pragma unroll
                for (int j = 0; j < 8; ++j) {
                    long grow = rowBase + mi * 16 + j;
                    if (grow >= M) continue;
                    long idx = grow * (long)N + gcol;
                    float v = acc[mi][ni][j] + bv;
                    if (MODE == 1) v = fmaxf(v, 0.0f);
                    if (MODE == 3) v = beta * v + ombeta * xold[idx];
                    outF[idx] = v;
                    if (MODE == 1 || MODE == 3) outBF[idx] = f2bf(v);
                }
            }
        }
    }
}

// ---------------- head-wise relation transform via WMMA ----------------
// Y[n][h*32+e] = sum_d X[n][h*32+d] * Wt[h][e][d]   (X f32, Wt bf16 packed)
__global__ __launch_bounds__(256) void head_transform_kernel(
    const float* __restrict__ X, const unsigned short* __restrict__ Wt,
    float* __restrict__ Y, int Nn) {
    const int h    = threadIdx.x >> 5;      // wave = head
    const int lane = threadIdx.x & 31;
    const int tile = blockIdx.x;            // 16-node tile
    const int lrow = lane & 15;
    const int khalf = (lane >> 4) << 3;
    const bool full = (tile * 16 + 16 <= Nn);

    Frag a;
    int node = tile * 16 + lrow;
    int nodeC = (node < Nn) ? node : (Nn - 1);      // clamp: garbage rows never stored
    {
        const float4* xp4 = (const float4*)(X + (long)nodeC * CDIM + h * DD + khalf);
        float4 f0 = xp4[0], f1 = xp4[1];            // K = khalf .. khalf+7
        float4 f2 = xp4[4], f3 = xp4[5];            // K = khalf+16 .. khalf+23
        a.h[0] = f2bf(f0.x);  a.h[1] = f2bf(f0.y);  a.h[2] = f2bf(f0.z);  a.h[3] = f2bf(f0.w);
        a.h[4] = f2bf(f1.x);  a.h[5] = f2bf(f1.y);  a.h[6] = f2bf(f1.z);  a.h[7] = f2bf(f1.w);
        a.h[8] = f2bf(f2.x);  a.h[9] = f2bf(f2.y);  a.h[10] = f2bf(f2.z); a.h[11] = f2bf(f2.w);
        a.h[12] = f2bf(f3.x); a.h[13] = f2bf(f3.y); a.h[14] = f2bf(f3.z); a.h[15] = f2bf(f3.w);
    }
    const unsigned short* wb = Wt + (long)h * 1024;
    Frag b0, b1;
    b0.u2[0] = *(const uint4*)&wb[(lrow)      * 32 + khalf];
    b0.u2[1] = *(const uint4*)&wb[(lrow)      * 32 + khalf + 16];
    b1.u2[0] = *(const uint4*)&wb[(16 + lrow) * 32 + khalf];
    b1.u2[1] = *(const uint4*)&wb[(16 + lrow) * 32 + khalf + 16];
    v8f acc0 = {}, acc1 = {};
    acc0 = wmma_bf16(a.v, b0.v, acc0);
    acc1 = wmma_bf16(a.v, b1.v, acc1);

    const int rbase = tile * 16 + ((lane >> 4) << 3);
    const int cbase = h * DD + lrow;
    if (full) {
#pragma unroll
        for (int j = 0; j < 8; ++j) {
            long nd = rbase + j;
            Y[nd * CDIM + cbase]      = acc0[j];
            Y[nd * CDIM + cbase + 16] = acc1[j];
        }
    } else {
#pragma unroll
        for (int j = 0; j < 8; ++j) {
            int nd = rbase + j;
            if (nd < Nn) {
                Y[(long)nd * CDIM + cbase]      = acc0[j];
                Y[(long)nd * CDIM + cbase + 16] = acc1[j];
            }
        }
    }
}

// ---------------- edge kernels (segment softmax + scatter) ----------------
__global__ void edge_alpha_max_kernel(const int* __restrict__ ei, int E,
                                      const float* __restrict__ q,
                                      const float* __restrict__ kr,
                                      const float* __restrict__ prel,
                                      float* __restrict__ alpha,
                                      unsigned* __restrict__ mEnc) {
    long gid = (long)blockIdx.x * blockDim.x + threadIdx.x;
    if (gid >= (long)E * HH) return;
    int h = (int)(gid & 7);
    long e = gid >> 3;
    int src = ei[e], dst = ei[(long)E + e];
    const float4* qp = (const float4*)(q  + (long)dst * CDIM + h * DD);
    const float4* kp = (const float4*)(kr + (long)src * CDIM + h * DD);
    float acc = 0.f;
#pragma unroll
    for (int j = 0; j < 8; ++j) {
        float4 a = qp[j], b = kp[j];
        acc += a.x * b.x + a.y * b.y + a.z * b.z + a.w * b.w;
    }
    float al = acc * prel[h] * SCALE;
    alpha[gid] = al;
    unsigned u = __float_as_uint(al);
    u = (u & 0x80000000u) ? ~u : (u | 0x80000000u);   // order-preserving encode
    atomicMax(&mEnc[(long)dst * HH + h], u);
}

__global__ void edge_exp_sum_kernel(const int* __restrict__ ei, int E,
                                    float* __restrict__ alpha,
                                    const unsigned* __restrict__ mEnc,
                                    float* __restrict__ sBuf) {
    long gid = (long)blockIdx.x * blockDim.x + threadIdx.x;
    if (gid >= (long)E * HH) return;
    int h = (int)(gid & 7);
    long e = gid >> 3;
    int dst = ei[(long)E + e];
    unsigned u = mEnc[(long)dst * HH + h];
    float m = (u == 0u) ? 0.0f
              : ((u >> 31) ? __uint_as_float(u & 0x7FFFFFFFu) : __uint_as_float(~u));
    float ex = __expf(alpha[gid] - m);
    alpha[gid] = ex;
    unsafeAtomicAdd(&sBuf[(long)dst * HH + h], ex);
}

__global__ void edge_scatter_kernel(const int* __restrict__ ei, int E,
                                    const float* __restrict__ alpha,
                                    const float* __restrict__ sBuf,
                                    const float* __restrict__ vr,
                                    float* __restrict__ agg) {
    long gid = (long)blockIdx.x * blockDim.x + threadIdx.x;
    if (gid >= (long)E * HH) return;
    int h = (int)(gid & 7);
    long e = gid >> 3;
    int src = ei[e], dst = ei[(long)E + e];
    float w = alpha[gid] / (sBuf[(long)dst * HH + h] + 1e-16f);
    const float4* vp = (const float4*)(vr + (long)src * CDIM + h * DD);
    float* ag = agg + (long)dst * CDIM + h * DD;
#pragma unroll
    for (int j = 0; j < 8; ++j) {
        float4 v = vp[j];
        unsafeAtomicAdd(&ag[j * 4 + 0], v.x * w);
        unsafeAtomicAdd(&ag[j * 4 + 1], v.y * w);
        unsafeAtomicAdd(&ag[j * 4 + 2], v.z * w);
        unsafeAtomicAdd(&ag[j * 4 + 3], v.w * w);
    }
}

// ---------------- host orchestration ----------------
extern "C" void kernel_launch(void* const* d_in, const int* in_sizes, int n_in,
                              void* d_out, int out_size, void* d_ws, size_t ws_size,
                              hipStream_t stream) {
    const float* x_author = (const float*)d_in[0];
    const float* x_paper  = (const float*)d_in[1];
    const float* liaW = (const float*)d_in[2];
    const float* liab = (const float*)d_in[3];
    const float* lipW = (const float*)d_in[4];
    const float* lipb = (const float*)d_in[5];
    const float* k_W = (const float*)d_in[6];
    const float* k_b = (const float*)d_in[7];
    const float* q_W = (const float*)d_in[8];
    const float* q_b = (const float*)d_in[9];
    const float* v_W = (const float*)d_in[10];
    const float* v_b = (const float*)d_in[11];
    const float* a_W = (const float*)d_in[12];
    const float* m_W = (const float*)d_in[13];
    const float* p_rel = (const float*)d_in[14];
    const float* skipArr = (const float*)d_in[15];
    const float* out_W = (const float*)d_in[16];
    const float* out_b = (const float*)d_in[17];
    const float* loW = (const float*)d_in[18];
    const float* lob = (const float*)d_in[19];
    const int* ei[3] = {(const int*)d_in[20], (const int*)d_in[21], (const int*)d_in[22]};
    int Ecnt[3] = {in_sizes[20] / 2, in_sizes[21] / 2, in_sizes[22] / 2};
    const int ES[3] = {0, 1, 1}, ED[3] = {1, 0, 1}, ER[3] = {0, 1, 2};
    const int NT[2] = {NA, NPP};

    // workspace carve-out
    char* cur = (char*)d_ws;
    auto carve = [&](size_t bytes) -> void* {
        void* r = (void*)cur;
        cur += (bytes + 255) & ~(size_t)255;
        return r;
    };
    unsigned short* wLinA = (unsigned short*)carve((size_t)256 * 128 * 2);
    unsigned short* wLinP = (unsigned short*)carve((size_t)256 * 256 * 2);
    unsigned short* wKQVO = (unsigned short*)carve((size_t)LL * TT * 4 * 65536 * 2);
    unsigned short* wLinOut = (unsigned short*)carve((size_t)OUTC * 256 * 2);
    unsigned short* wA = (unsigned short*)carve((size_t)LL * RR * HH * 1024 * 2);
    unsigned short* wM = (unsigned short*)carve((size_t)LL * RR * HH * 1024 * 2);
    unsigned short* xaBF = (unsigned short*)carve((size_t)NA * FA * 2);
    unsigned short* xpBF = (unsigned short*)carve((size_t)NPP * FP * 2);
    float* xsF[2];  unsigned short* xsBF[2];
    float* kF[2];   float* qF[2];  float* vF[2];
    float* aggF[2]; unsigned short* aggBF[2];
    for (int t = 0; t < 2; ++t) {
        size_t n = (size_t)NT[t] * CDIM;
        xsF[t]  = (float*)carve(n * 4);
        xsBF[t] = (unsigned short*)carve(n * 2);
        kF[t] = (float*)carve(n * 4);
        qF[t] = (float*)carve(n * 4);
        vF[t] = (float*)carve(n * 4);
        aggF[t]  = (float*)carve(n * 4);
        aggBF[t] = (unsigned short*)carve(n * 2);
    }
    float* krF = (float*)carve((size_t)NPP * CDIM * 4);
    float* vrF = (float*)carve((size_t)NPP * CDIM * 4);
    int maxE = Ecnt[0];
    if (Ecnt[1] > maxE) maxE = Ecnt[1];
    if (Ecnt[2] > maxE) maxE = Ecnt[2];
    float* alphaBuf = (float*)carve((size_t)maxE * HH * 4);
    unsigned* mEnc = (unsigned*)carve((size_t)NPP * HH * 4);
    float* sBuf = (float*)carve((size_t)NPP * HH * 4);

    auto grid1 = [](long n) { return dim3((unsigned)((n + 255) / 256)); };
    dim3 blk(256);

    auto gemm = [&](int mode, const unsigned short* A, const unsigned short* Bt,
                    const float* bias, int M, int N, int K,
                    float* oF, unsigned short* oBF,
                    const float* xold, int skipIdx) {
        dim3 g((N + BN - 1) / BN, (M + BM - 1) / BM);
        if (mode == 1)
            wmma_gemm_kernel<1><<<g, blk, 0, stream>>>(A, Bt, bias, M, N, K, oF, oBF, nullptr, nullptr, 0);
        else if (mode == 2)
            wmma_gemm_kernel<2><<<g, blk, 0, stream>>>(A, Bt, bias, M, N, K, oF, nullptr, nullptr, nullptr, 0);
        else
            wmma_gemm_kernel<3><<<g, blk, 0, stream>>>(A, Bt, bias, M, N, K, oF, oBF, xold, skipArr, skipIdx);
    };

    // ---- weight packing (bf16, [N][K]) ----
    pack_wt_kernel<<<grid1((long)128 * 256), blk, 0, stream>>>(liaW, wLinA, 128, 256);
    pack_wt_kernel<<<grid1((long)256 * 256), blk, 0, stream>>>(lipW, wLinP, 256, 256);
    for (int l = 0; l < LL; ++l)
        for (int t = 0; t < TT; ++t) {
            size_t off = (size_t)(l * TT + t) * 65536;
            size_t pk = ((size_t)(l * TT + t) * 4) * 65536;
            pack_wt_kernel<<<grid1(65536), blk, 0, stream>>>(k_W + off, wKQVO + pk + 0 * 65536, 256, 256);
            pack_wt_kernel<<<grid1(65536), blk, 0, stream>>>(q_W + off, wKQVO + pk + 1 * 65536, 256, 256);
            pack_wt_kernel<<<grid1(65536), blk, 0, stream>>>(v_W + off, wKQVO + pk + 2 * 65536, 256, 256);
            pack_wt_kernel<<<grid1(65536), blk, 0, stream>>>(out_W + off, wKQVO + pk + 3 * 65536, 256, 256);
        }
    pack_wt_kernel<<<grid1((long)256 * OUTC), blk, 0, stream>>>(loW, wLinOut, 256, OUTC);
    pack_head_kernel<<<grid1((long)LL * RR * HH * 1024), blk, 0, stream>>>(a_W, wA, LL * RR * HH);
    pack_head_kernel<<<grid1((long)LL * RR * HH * 1024), blk, 0, stream>>>(m_W, wM, LL * RR * HH);

    // ---- input projection: xs[t] = relu(x @ W + b) ----
    cvt_bf16_kernel<<<grid1((long)NA * FA), blk, 0, stream>>>(x_author, xaBF, (long)NA * FA);
    cvt_bf16_kernel<<<grid1((long)NPP * FP), blk, 0, stream>>>(x_paper, xpBF, (long)NPP * FP);
    gemm(1, xaBF, wLinA, liab, NA, CDIM, FA, xsF[0], xsBF[0], nullptr, 0);
    gemm(1, xpBF, wLinP, lipb, NPP, CDIM, FP, xsF[1], xsBF[1], nullptr, 0);

    for (int l = 0; l < LL; ++l) {
        // k/q/v projections
        for (int t = 0; t < TT; ++t) {
            size_t pk = ((size_t)(l * TT + t) * 4) * 65536;
            const float* kb = k_b + (size_t)(l * TT + t) * CDIM;
            const float* qb = q_b + (size_t)(l * TT + t) * CDIM;
            const float* vb = v_b + (size_t)(l * TT + t) * CDIM;
            gemm(2, xsBF[t], wKQVO + pk + 0 * 65536, kb, NT[t], CDIM, CDIM, kF[t], nullptr, nullptr, 0);
            gemm(2, xsBF[t], wKQVO + pk + 1 * 65536, qb, NT[t], CDIM, CDIM, qF[t], nullptr, nullptr, 0);
            gemm(2, xsBF[t], wKQVO + pk + 2 * 65536, vb, NT[t], CDIM, CDIM, vF[t], nullptr, nullptr, 0);
        }
        (void)hipMemsetAsync(aggF[0], 0, (size_t)NA * CDIM * 4, stream);
        (void)hipMemsetAsync(aggF[1], 0, (size_t)NPP * CDIM * 4, stream);

        for (int rel = 0; rel < RR; ++rel) {
            int s = ES[rel], d = ED[rel], r = ER[rel];
            int E = Ecnt[rel];
            const unsigned short* aW = wA + (size_t)(l * RR + r) * HH * 1024;
            const unsigned short* mW = wM + (size_t)(l * RR + r) * HH * 1024;
            head_transform_kernel<<<(NT[s] + 15) / 16, blk, 0, stream>>>(kF[s], aW, krF, NT[s]);
            head_transform_kernel<<<(NT[s] + 15) / 16, blk, 0, stream>>>(vF[s], mW, vrF, NT[s]);
            (void)hipMemsetAsync(mEnc, 0, (size_t)NT[d] * HH * 4, stream);
            (void)hipMemsetAsync(sBuf, 0, (size_t)NT[d] * HH * 4, stream);
            const float* prel = p_rel + (size_t)(l * RR + r) * HH;
            long tot = (long)E * HH;
            edge_alpha_max_kernel<<<grid1(tot), blk, 0, stream>>>(ei[rel], E, qF[d], krF, prel, alphaBuf, mEnc);
            edge_exp_sum_kernel<<<grid1(tot), blk, 0, stream>>>(ei[rel], E, alphaBuf, mEnc, sBuf);
            edge_scatter_kernel<<<grid1(tot), blk, 0, stream>>>(ei[rel], E, alphaBuf, sBuf, vrF, aggF[d]);
        }

        // out projection + skip blend
        for (int t = 0; t < TT; ++t) {
            size_t pk = ((size_t)(l * TT + t) * 4) * 65536;
            const float* ob = out_b + (size_t)(l * TT + t) * CDIM;
            long n = (long)NT[t] * CDIM;
            gelu_to_bf16_kernel<<<grid1(n), blk, 0, stream>>>(aggF[t], aggBF[t], n);
            gemm(3, aggBF[t], wKQVO + pk + 3 * 65536, ob, NT[t], CDIM, CDIM,
                 xsF[t], xsBF[t], xsF[t], l * TT + t);
        }
    }

    // ---- final shared linear ----
    float* outF = (float*)d_out;
    gemm(2, xsBF[0], wLinOut, lob, NA, OUTC, CDIM, outF, nullptr, nullptr, 0);
    gemm(2, xsBF[1], wLinOut, lob, NPP, OUTC, CDIM, outF + (size_t)NA * OUTC, nullptr, nullptr, 0);
}
